// MultiHeadAttentionModify_4999341932596
// MI455X (gfx1250) — matrix operand order repaired and checked by reading, not compiled, not run on hardware
//
#include <hip/hip_runtime.h>
#include <math.h>

typedef __bf16 bf16;
typedef __attribute__((ext_vector_type(16))) __bf16 v16bf;
typedef __attribute__((ext_vector_type(8)))  float  v8f;

#define D   256
#define NQc 4096
#define NKc 4096

union FragBits { v16bf v; int4 q[2]; };

static __device__ __forceinline__ v8f wmma_bf16(v16bf a, v16bf b, v8f c) {
  // (neg_a, A, neg_b, B, c_mod, C, reuse_a, reuse_b)
  return __builtin_amdgcn_wmma_f32_16x16x32_bf16(false, a, false, b, (short)0, c, false, false);
}

// ---- async global->LDS staging (ASYNCcnt path, ISA 15.18.3) -------------
static __device__ __forceinline__ void async_b128(unsigned lds_addr, const void* gaddr) {
  asm volatile("global_load_async_to_lds_b128 %0, %1, off"
               :: "v"(lds_addr), "v"(gaddr) : "memory");
}
static __device__ __forceinline__ void wait_asynccnt0() {
  asm volatile("s_wait_asynccnt 0" ::: "memory");
}

// ---- fragment loaders ---------------------------------------------------
// A (16xK=32), row-major bf16 source: lane l<16 -> row l, K {k0..k0+7, k0+16..k0+23}
//                                     lane l>=16 -> row l-16, K {k0+8..15, k0+24..31}
static __device__ __forceinline__ v16bf load_fragA_bf16(const bf16* base, int ld,
                                                        int row0, int k0, int lane) {
  const int r = lane & 15, h = lane >> 4;
  const bf16* p = base + (size_t)(row0 + r) * ld + k0 + h * 8;
  FragBits f;
  f.q[0] = *(const int4*)p;
  f.q[1] = *(const int4*)(p + 16);
  return f.v;
}
// B (K=32 x 16), column n = row n of a row-major Y (X·Yt pattern):
// lane n<16 -> col n, K {k0..k0+7, k0+16..k0+23} ; lane n+16 -> col n, K {k0+8..15, k0+24..31}
static __device__ __forceinline__ v16bf load_fragB_bf16(const bf16* base, int ld,
                                                        int col0, int k0, int lane) {
  const int c = lane & 15, h = lane >> 4;
  const bf16* p = base + (size_t)(col0 + c) * ld + k0 + h * 8;
  FragBits f;
  f.q[0] = *(const int4*)p;
  f.q[1] = *(const int4*)(p + 16);
  return f.v;
}
// same, but fp32 source converted to bf16 on the fly
static __device__ __forceinline__ v16bf load_fragA_f32(const float* base, int ld,
                                                       int row0, int k0, int lane) {
  const int r = lane & 15, h = lane >> 4;
  const float* p = base + (size_t)(row0 + r) * ld + k0 + h * 8;
  v16bf f;
#pragma unroll
  for (int i = 0; i < 8; ++i) { f[i] = (bf16)p[i]; f[i + 8] = (bf16)p[i + 16]; }
  return f;
}
// conv A fragment: rows are pixels p0+r shifted by (dy,dx) with zero padding
static __device__ __forceinline__ v16bf load_fragA_conv(const float* xb, int p0,
                                                        int dy, int dx, int k0, int lane) {
  const int r = lane & 15, h = lane >> 4;
  const int p = p0 + r;
  const int y = (p >> 6) + dy, x = (p & 63) + dx;
  v16bf f;
  if ((unsigned)y < 64u && (unsigned)x < 64u) {
    const float* q = xb + (size_t)((y << 6) + x) * D + k0 + h * 8;
#pragma unroll
    for (int i = 0; i < 8; ++i) { f[i] = (bf16)q[i]; f[i + 8] = (bf16)q[i + 16]; }
  } else {
#pragma unroll
    for (int i = 0; i < 16; ++i) f[i] = (bf16)0.0f;
  }
  return f;
}

// ---- K0: weight folding / conversion ------------------------------------
__global__ void k_fold(const float* __restrict__ w1, const float* __restrict__ w2,
                       const float* __restrict__ wvs_f, const float* __restrict__ w2c_f,
                       const float* __restrict__ c1w,
                       const float* __restrict__ g, const float* __restrict__ be,
                       const float* __restrict__ mu, const float* __restrict__ va,
                       bf16* __restrict__ w1f, bf16* __restrict__ w2f,
                       bf16* __restrict__ wvs, bf16* __restrict__ w2c,
                       bf16* __restrict__ wk, float* __restrict__ bna,
                       float* __restrict__ bns) {
  int i = blockIdx.x * 256 + threadIdx.x;
  if (i < 65536) {                       // fold concat: W1eff = W1[:, :256] + W1[:, 256:]
    int h = i >> 8, d = i & 255;
    w1f[i] = (bf16)(w1[h * 512 + d] + w1[h * 512 + 256 + d]);
  } else if (i < 131072) {
    int j = i - 65536;  w2f[j] = (bf16)w2[j];
  } else if (i < 196608) {
    int j = i - 131072; wvs[j] = (bf16)wvs_f[j];
  } else if (i < 262144) {
    int j = i - 196608; w2c[j] = (bf16)w2c_f[j];
  } else if (i < 262144 + 589824) {      // conv1 [oc][ic][3][3] -> [kk][oc][ic]
    int j = i - 262144;
    int kk = j >> 16, oi = j & 65535;
    wk[j] = (bf16)c1w[oi * 9 + kk];
  } else if (i < 262144 + 589824 + 256) { // BN fold
    int d = i - (262144 + 589824);
    float a = g[d] / sqrtf(va[d] + 1e-3f);
    bna[d] = a;  bns[d] = be[d] - mu[d] * a;
  }
}

// ---- K1: L2-normalize key_pos -> bf16 -----------------------------------
__global__ void k_knorm(const float* __restrict__ kp, bf16* __restrict__ kn) {
  int tid = threadIdx.x;
  int row = blockIdx.x * 32 + (tid >> 3);
  int c0 = (tid & 7) * 32;
  const float* p = kp + (size_t)row * D + c0;
  float va[32]; float ss = 0.f;
#pragma unroll
  for (int i = 0; i < 32; ++i) { va[i] = p[i]; ss += va[i] * va[i]; }
  ss += __shfl_xor(ss, 1, 32);
  ss += __shfl_xor(ss, 2, 32);
  ss += __shfl_xor(ss, 4, 32);
  float inv = 1.f / sqrtf(ss);
  bf16* o = kn + (size_t)row * D + c0;
#pragma unroll
  for (int i = 0; i < 32; ++i) o[i] = (bf16)(va[i] * inv);
}

// ---- K2: v projection * other_W, stored transposed vT[b][d][k] bf16 -----
__global__ void k_vproj(const float* __restrict__ v, const float* __restrict__ oW,
                        const bf16* __restrict__ wvs, bf16* __restrict__ vT) {
  int b = blockIdx.y;
  int wave = threadIdx.x >> 5, lane = threadIdx.x & 31;
  int k0 = blockIdx.x * 16;
  int n = lane & 15, hi = (lane >> 4) << 3;
  const float* vb = v + ((size_t)b * NKc + k0) * D;
  v16bf av[8];
#pragma unroll
  for (int ec = 0; ec < 8; ++ec) av[ec] = load_fragA_f32(vb, D, 0, ec * 32, lane);
  int dt0 = wave * 2;
  v8f acc[2] = {};
#pragma unroll
  for (int ec = 0; ec < 8; ++ec) {
    acc[0] = wmma_bf16(av[ec], load_fragB_bf16(wvs, D, dt0 * 16, ec * 32, lane), acc[0]);
    acc[1] = wmma_bf16(av[ec], load_fragB_bf16(wvs, D, (dt0 + 1) * 16, ec * 32, lane), acc[1]);
  }
  float w8[8];
#pragma unroll
  for (int r = 0; r < 8; ++r) w8[r] = oW[(size_t)b * NKc + k0 + hi + r];
#pragma unroll
  for (int t = 0; t < 2; ++t) {
    int dcol = (dt0 + t) * 16 + n;
    union { bf16 h[8]; int4 q; } pk;
#pragma unroll
    for (int r = 0; r < 8; ++r) pk.h[r] = (bf16)(acc[t][r] * w8[r]);
    *(int4*)(vT + ((size_t)b * D + dcol) * (size_t)NKc + k0 + hi) = pk.q;
  }
}

// ---- K3: ref_point_head MLP (folded) + L2 norm -> qn bf16 ---------------
__global__ void k_qmlp(const float* __restrict__ qpos, const bf16* __restrict__ w1f,
                       const float* __restrict__ b1, const bf16* __restrict__ w2f,
                       const float* __restrict__ b2, bf16* __restrict__ qn) {
  __shared__ bf16 hid[8][16][D];         // 64 KB
  int b = blockIdx.y;
  int wave = threadIdx.x >> 5, lane = threadIdx.x & 31;
  int q0 = blockIdx.x * 128 + wave * 16;
  int n = lane & 15, hi = (lane >> 4) << 3;
  const float* qpb = qpos + ((size_t)b * NQc + q0) * D;

  // phase 1: hidden = relu(qpos @ W1eff^T + b1)
  v16bf aq[8];
#pragma unroll
  for (int ec = 0; ec < 8; ++ec) aq[ec] = load_fragA_f32(qpb, D, 0, ec * 32, lane);
  for (int t = 0; t < 16; ++t) {
    v8f acc = {};
#pragma unroll
    for (int ec = 0; ec < 8; ++ec)
      acc = wmma_bf16(aq[ec], load_fragB_bf16(w1f, D, t * 16, ec * 32, lane), acc);
    float bb = b1[t * 16 + n];
#pragma unroll
    for (int r = 0; r < 8; ++r) {
      float x = acc[r] + bb;
      hid[wave][hi + r][t * 16 + n] = (bf16)(x > 0.f ? x : 0.f);
    }
  }
  __syncthreads();

  // phase 2: out = hidden @ W2^T + b2 ; accumulate row sumsq; stash unscaled
  v16bf ah[8];
#pragma unroll
  for (int ec = 0; ec < 8; ++ec)
    ah[ec] = load_fragA_bf16(&hid[wave][0][0], D, 0, ec * 32, lane);
  float sq[8] = {};
  for (int t = 0; t < 16; ++t) {
    v8f acc = {};
#pragma unroll
    for (int ec = 0; ec < 8; ++ec)
      acc = wmma_bf16(ah[ec], load_fragB_bf16(w2f, D, t * 16, ec * 32, lane), acc);
    float bb = b2[t * 16 + n];
#pragma unroll
    for (int r = 0; r < 8; ++r) {
      float x = acc[r] + bb;
      sq[r] += x * x;
      hid[wave][hi + r][t * 16 + n] = (bf16)x;
    }
  }
#pragma unroll
  for (int m = 1; m <= 8; m <<= 1) {
#pragma unroll
    for (int r = 0; r < 8; ++r) sq[r] += __shfl_xor(sq[r], m, 32);
  }
  float inv[8];
#pragma unroll
  for (int r = 0; r < 8; ++r) inv[r] = 1.f / sqrtf(sq[r]);
  bf16* qnb = qn + ((size_t)b * NQc + q0) * D;
  for (int t = 0; t < 16; ++t) {
#pragma unroll
    for (int r = 0; r < 8; ++r)
      qnb[(size_t)(hi + r) * D + t * 16 + n] =
          (bf16)((float)hid[wave][hi + r][t * 16 + n] * inv[r]);
  }
}

// ---- K4: fused attention: out = relu(qn·kn^T) · vT^T  (flash-style) -----
// kn / vT tiles staged into LDS with async-to-LDS copies, double-buffered;
// all 8 waves share the staged tiles (removes 4x redundant global reads).
__global__ void k_attn(const bf16* __restrict__ qn, const bf16* __restrict__ kn,
                       const bf16* __restrict__ vT, float* __restrict__ xa) {
  __shared__ bf16 KT[2][32][D];          // keys tile: 32 rows x 256 e (2 x 16 KB)
  __shared__ bf16 VTl[2][D][32];         // v tile:   256 d   x 32 k  (2 x 16 KB)
  __shared__ bf16 S[4][16][32];          // relu(S) tiles (4 KB)
  int b = blockIdx.y;
  int tid = threadIdx.x;
  int wave = tid >> 5, lane = tid & 31;
  int qs = wave >> 1, kh = wave & 1;
  int q0 = blockIdx.x * 64 + qs * 16;
  int n = lane & 15, hi = (lane >> 4) << 3;
  const bf16* qnb = qn + (size_t)b * NQc * D;
  const bf16* knb = kn + (size_t)b * NKc * D;
  const bf16* vtb = vT + (size_t)b * D * (size_t)NKc;

  v16bf aq[8];
#pragma unroll
  for (int ec = 0; ec < 8; ++ec) aq[ec] = load_fragA_bf16(qnb, D, q0, ec * 32, lane);

  // per-thread staging assignment: 64 B of KT + 64 B of VT per chunk
  const int krow = tid >> 3, kcol = (tid & 7) * 32;
  const bf16* gK = knb + (size_t)krow * D + kcol;        // advances by 32*D per chunk
  const bf16* gV = vtb + (size_t)tid * (size_t)NKc;      // advances by 32 per chunk
  unsigned uK[2] = { (unsigned)(uintptr_t)&KT[0][krow][kcol],
                     (unsigned)(uintptr_t)&KT[1][krow][kcol] };
  unsigned uV[2] = { (unsigned)(uintptr_t)&VTl[0][tid][0],
                     (unsigned)(uintptr_t)&VTl[1][tid][0] };

  // prologue: stage chunk 0 into buffer 0
#pragma unroll
  for (int i = 0; i < 4; ++i) async_b128(uK[0] + i * 16, (const char*)gK + i * 16);
#pragma unroll
  for (int i = 0; i < 4; ++i) async_b128(uV[0] + i * 16, (const char*)gV + i * 16);
  gK += 32 * D;  gV += 32;

  v8f acc[8] = {};
  int dbase = kh * 128;
  int cur = 0;

  for (int kc = 0; kc < 128; ++kc) {
    int k0 = kc * 32;
    // staged data for this chunk is complete after own-wait + barrier;
    // this barrier also fences iter kc-1's S reads from iter kc's S writes.
    wait_asynccnt0();
    __syncthreads();

    // S tile: 16 q-rows x 16 keys (this wave's half of the 32-key chunk), from LDS
    v8f s = {};
#pragma unroll
    for (int ec = 0; ec < 8; ++ec)
      s = wmma_bf16(aq[ec], load_fragB_bf16(&KT[cur][0][0], D, kh * 16, ec * 32, lane), s);

    // overlap: stage next chunk into the other buffer (no reader of it yet;
    // re-stage of THIS buffer only happens after the next loop-top barrier)
    if (kc + 1 < 128) {
#pragma unroll
      for (int i = 0; i < 4; ++i) async_b128(uK[cur ^ 1] + i * 16, (const char*)gK + i * 16);
#pragma unroll
      for (int i = 0; i < 4; ++i) async_b128(uV[cur ^ 1] + i * 16, (const char*)gV + i * 16);
      gK += 32 * D;  gV += 32;
    }

    bf16* sp = &S[qs][0][0];
#pragma unroll
    for (int r = 0; r < 8; ++r) {
      float x = s[r];
      sp[(hi + r) * 32 + kh * 16 + n] = (bf16)(x > 0.f ? x : 0.f);
    }
    __syncthreads();

    // second GEMM: acc[q, d] += relu(S)[q, 32k] * vT[d, 32k]^T  (all from LDS)
    v16bf sa = load_fragA_bf16(&S[qs][0][0], 32, 0, 0, lane);
#pragma unroll
    for (int dt = 0; dt < 8; ++dt)
      acc[dt] = wmma_bf16(sa, load_fragB_bf16(&VTl[cur][0][0], 32, dbase + dt * 16, 0, lane),
                          acc[dt]);
    cur ^= 1;
    (void)k0;
  }

  float* xab = xa + ((size_t)b * NQc + q0) * D;   // channels-last [b][p][d]
#pragma unroll
  for (int dt = 0; dt < 8; ++dt) {
#pragma unroll
    for (int r = 0; r < 8; ++r)
      xab[(size_t)(hi + r) * D + dbase + dt * 16 + n] = acc[dt][r];
  }
}

// ---- K5: conv3x3 (9 shifted GEMMs) + BN + SiLU, channels-last -----------
__global__ void k_conv1(const float* __restrict__ xa, const bf16* __restrict__ wk,
                        const float* __restrict__ bna, const float* __restrict__ bns,
                        bf16* __restrict__ c1) {
  int b = blockIdx.y;
  int wave = threadIdx.x >> 5, lane = threadIdx.x & 31;
  int ps = wave >> 1, kh = wave & 1;
  int p0 = blockIdx.x * 64 + ps * 16;
  int n = lane & 15, hi = (lane >> 4) << 3;
  const float* xb = xa + (size_t)b * NQc * D;
  int ocb = kh * 128;
  v8f acc[8] = {};
  for (int kk = 0; kk < 9; ++kk) {
    int dy = kk / 3 - 1, dx = kk % 3 - 1;
    const bf16* wkk = wk + (size_t)kk * 65536;
#pragma unroll
    for (int ec = 0; ec < 8; ++ec) {
      v16bf a = load_fragA_conv(xb, p0, dy, dx, ec * 32, lane);
#pragma unroll
      for (int ot = 0; ot < 8; ++ot)
        acc[ot] = wmma_bf16(a, load_fragB_bf16(wkk, D, ocb + ot * 16, ec * 32, lane), acc[ot]);
    }
  }
  bf16* c1b = c1 + ((size_t)b * NQc + p0) * D;
#pragma unroll
  for (int ot = 0; ot < 8; ++ot) {
    int oc = ocb + ot * 16 + n;
    float a = bna[oc], sh = bns[oc];
#pragma unroll
    for (int r = 0; r < 8; ++r) {
      float y = acc[ot][r] * a + sh;
      y = y * (1.f / (1.f + __expf(-y)));          // SiLU
      c1b[(size_t)(hi + r) * D + oc] = (bf16)y;
    }
  }
}

// ---- K6: conv1x1 + bias + residual, store NCHW --------------------------
__global__ void k_conv2(const bf16* __restrict__ c1, const bf16* __restrict__ w2c,
                        const float* __restrict__ cb, const float* __restrict__ query,
                        float* __restrict__ out) {
  int b = blockIdx.y;
  int wave = threadIdx.x >> 5, lane = threadIdx.x & 31;
  int p0 = blockIdx.x * 16;
  int n = lane & 15, hi = (lane >> 4) << 3;
  const bf16* c1b = c1 + ((size_t)b * NQc + p0) * D;
  v16bf ac[8];
#pragma unroll
  for (int ec = 0; ec < 8; ++ec) ac[ec] = load_fragA_bf16(c1b, D, 0, ec * 32, lane);
  int dt0 = wave * 2;
  v8f aa[2] = {};
#pragma unroll
  for (int ec = 0; ec < 8; ++ec) {
    aa[0] = wmma_bf16(ac[ec], load_fragB_bf16(w2c, D, dt0 * 16, ec * 32, lane), aa[0]);
    aa[1] = wmma_bf16(ac[ec], load_fragB_bf16(w2c, D, (dt0 + 1) * 16, ec * 32, lane), aa[1]);
  }
#pragma unroll
  for (int t = 0; t < 2; ++t) {
    int oc = (dt0 + t) * 16 + n;
    float bias = cb[oc];
    size_t base = ((size_t)b * D + oc) * (size_t)NQc + p0 + hi;  // p contiguous per lane
    const float* qp = query + base;
    float* op = out + base;
    float4 qlo = *(const float4*)qp;
    float4 qhi2 = *(const float4*)(qp + 4);
    float4 olo = make_float4(aa[t][0] + bias + qlo.x,  aa[t][1] + bias + qlo.y,
                             aa[t][2] + bias + qlo.z,  aa[t][3] + bias + qlo.w);
    float4 ohi = make_float4(aa[t][4] + bias + qhi2.x, aa[t][5] + bias + qhi2.y,
                             aa[t][6] + bias + qhi2.z, aa[t][7] + bias + qhi2.w);
    *(float4*)op = olo;
    *(float4*)(op + 4) = ohi;
  }
}

// ---- launch --------------------------------------------------------------
extern "C" void kernel_launch(void* const* d_in, const int* in_sizes, int n_in,
                              void* d_out, int out_size, void* d_ws, size_t ws_size,
                              hipStream_t stream) {
  const float* query     = (const float*)d_in[0];
  const float* v_in      = (const float*)d_in[1];
  const float* query_pos = (const float*)d_in[2];
  const float* key_pos   = (const float*)d_in[3];
  const float* other_W   = (const float*)d_in[4];
  const float* w_vs      = (const float*)d_in[5];
  const float* mlp_w1    = (const float*)d_in[6];
  const float* mlp_b1    = (const float*)d_in[7];
  const float* mlp_w2    = (const float*)d_in[8];
  const float* mlp_b2    = (const float*)d_in[9];
  const float* conv1_w   = (const float*)d_in[10];
  const float* bn_g      = (const float*)d_in[11];
  const float* bn_b      = (const float*)d_in[12];
  const float* bn_m      = (const float*)d_in[13];
  const float* bn_v      = (const float*)d_in[14];
  const float* conv2_w   = (const float*)d_in[15];
  const float* conv2_b   = (const float*)d_in[16];
  float* out = (float*)d_out;

  char* ws = (char*)d_ws;
  const size_t MB = 1u << 20;
  bf16*  vT  = (bf16*)(ws + 0);            // [4][256][4096] bf16  (8 MB)
  bf16*  qn  = (bf16*)(ws + 8 * MB);       // [4][4096][256] bf16  (8 MB)
  bf16*  kn  = (bf16*)(ws + 16 * MB);      // [4][4096][256] bf16  (8 MB)
  float* xa  = (float*)(ws + 24 * MB);     // [4][4096][256] f32   (16 MB)
  bf16*  c1  = (bf16*)(ws + 40 * MB);      // [4][4096][256] bf16  (8 MB)
  bf16*  w1f = (bf16*)(ws + 48 * MB);                 // 128 KB
  bf16*  w2f = (bf16*)(ws + 48 * MB + 131072);        // 128 KB
  bf16*  wvs = (bf16*)(ws + 48 * MB + 262144);        // 128 KB
  bf16*  w2c = (bf16*)(ws + 48 * MB + 393216);        // 128 KB
  bf16*  wk  = (bf16*)(ws + 48 * MB + 524288);        // 1.125 MB
  float* bna = (float*)(ws + 48 * MB + 524288 + 1179648);
  float* bns = (float*)(ws + 48 * MB + 524288 + 1179648 + 1024);

  k_fold <<<3329, 256, 0, stream>>>(mlp_w1, mlp_w2, w_vs, conv2_w, conv1_w,
                                    bn_g, bn_b, bn_m, bn_v,
                                    w1f, w2f, wvs, w2c, wk, bna, bns);
  k_knorm<<<512, 256, 0, stream>>>(key_pos, kn);
  k_vproj<<<dim3(NKc / 16, 4), 256, 0, stream>>>(v_in, other_W, wvs, vT);
  k_qmlp <<<dim3(NQc / 128, 4), 256, 0, stream>>>(query_pos, w1f, mlp_b1, w2f, mlp_b2, qn);
  k_attn <<<dim3(NQc / 64, 4), 256, 0, stream>>>(qn, kn, vT, xa);
  k_conv1<<<dim3(NQc / 64, 4), 256, 0, stream>>>(xa, wk, bna, bns, c1);
  k_conv2<<<dim3(NQc / 16, 4), 256, 0, stream>>>(c1, w2c, conv2_b, query, out);
}
